// T1Layer_37271726195188
// MI455X (gfx1250) — matrix operand order repaired and checked by reading, not compiled, not run on hardware
//
#include <hip/hip_runtime.h>

#define NN    100000
#define EE    1000000
#define DD    128
#define AGGD  129
#define OUTD  257
#define BN_EPS 1e-5f

#define MTILES   (NN / 16)      // 6250, exact
#define K1P      160            // gemm1 K padded (129 -> 5*32)
#define K2P      288            // gemm2 K padded (257 -> 9*32)
#define W1ROWS   192            // w1 packed rows (3 groups * 64)
#define W2ROWS   320            // w2 packed rows (5 groups * 64)

typedef __attribute__((ext_vector_type(16))) __bf16 v16bf;
typedef __attribute__((ext_vector_type(8)))  __bf16 v8bf;
typedef __attribute__((ext_vector_type(8)))  float  v8f;

__device__ __forceinline__ v16bf cat8(v8bf a, v8bf b) {
    return __builtin_shufflevector(a, b, 0,1,2,3,4,5,6,7,8,9,10,11,12,13,14,15);
}

// ---------------------------------------------------------------- zero
__global__ void zero_kernel(float* __restrict__ p, long long n) {
    long long i = (long long)blockIdx.x * blockDim.x + threadIdx.x;
    long long stride = (long long)gridDim.x * blockDim.x;
    for (; i < n; i += stride) p[i] = 0.0f;
}

// ---------------------------------------------------------------- scatter
// One wave (32 lanes) per edge. Each lane handles 4 columns via float4.
__global__ void scatter_kernel(const int* __restrict__ u, const int* __restrict__ v,
                               const float* __restrict__ g,
                               const float* __restrict__ ru, const float* __restrict__ rv,
                               float* __restrict__ agg) {
    int e = blockIdx.x * (blockDim.x >> 5) + (threadIdx.x >> 5);
    if (e >= EE) return;
    int lane = threadIdx.x & 31;
    int uu = u[e];
    int vv = v[e];
    float ge = g[e];

    float4 a = ((const float4*)(ru + (size_t)e * DD))[lane];
    float* dv = agg + (size_t)vv * AGGD + lane * 4;
    unsafeAtomicAdd(dv + 0, a.x);
    unsafeAtomicAdd(dv + 1, a.y);
    unsafeAtomicAdd(dv + 2, a.z);
    unsafeAtomicAdd(dv + 3, a.w);

    float4 b = ((const float4*)(rv + (size_t)e * DD))[lane];
    float* du = agg + (size_t)uu * AGGD + lane * 4;
    unsafeAtomicAdd(du + 0, b.x);
    unsafeAtomicAdd(du + 1, b.y);
    unsafeAtomicAdd(du + 2, b.z);
    unsafeAtomicAdd(du + 3, b.w);

    if (lane == 0) {
        unsafeAtomicAdd(agg + (size_t)vv * AGGD + DD, ge);
        unsafeAtomicAdd(agg + (size_t)uu * AGGD + DD, ge);
    }
}

// ---------------------------------------------------------------- column stats
#define STAT_BLOCKS 400
#define ROWS_PER_BLOCK 250
__global__ void colstats_kernel(const float* __restrict__ agg,
                                float* __restrict__ sum, float* __restrict__ sumsq) {
    int c = threadIdx.x;
    if (c >= AGGD) return;
    int r0 = blockIdx.x * ROWS_PER_BLOCK;
    int r1 = r0 + ROWS_PER_BLOCK; if (r1 > NN) r1 = NN;
    float s = 0.0f, ss = 0.0f;
    for (int r = r0; r < r1; ++r) {
        float x = agg[(size_t)r * AGGD + c];
        s += x; ss += x * x;
    }
    unsafeAtomicAdd(&sum[c], s);
    unsafeAtomicAdd(&sumsq[c], ss);
}

__global__ void finalize_kernel(const float* __restrict__ sum, const float* __restrict__ sumsq,
                                const float* __restrict__ bnw, const float* __restrict__ bnb,
                                float* __restrict__ scale, float* __restrict__ shift) {
    int c = threadIdx.x;
    if (c >= AGGD) return;
    float mean = sum[c] * (1.0f / NN);
    float var  = sumsq[c] * (1.0f / NN) - mean * mean;
    float inv  = rsqrtf(var + BN_EPS);
    float sc   = bnw[c] * inv;
    scale[c] = sc;
    shift[c] = bnb[c] - mean * sc;
}

// ---------------------------------------------------------------- packing
// BN-apply and split x into hi/lo bf16, padded N x 160 (pad cols = 0).
__global__ void bn_pack_kernel(const float* __restrict__ agg,
                               const float* __restrict__ scale, const float* __restrict__ shift,
                               __bf16* __restrict__ xph, __bf16* __restrict__ xpl) {
    int c = threadIdx.x;                      // 0..159
    float sc = (c < AGGD) ? scale[c] : 0.0f;
    float sh = (c < AGGD) ? shift[c] : 0.0f;
    int r0 = blockIdx.x * ROWS_PER_BLOCK;
    int r1 = r0 + ROWS_PER_BLOCK; if (r1 > NN) r1 = NN;
    for (int r = r0; r < r1; ++r) {
        float v = (c < AGGD) ? (agg[(size_t)r * AGGD + c] * sc + sh) : 0.0f;
        __bf16 hi = (__bf16)v;
        size_t idx = (size_t)r * K1P + c;
        xph[idx] = hi;
        xpl[idx] = (__bf16)(v - (float)hi);
    }
}

// Pack h (cols 0..127) of Z = [h | h1] and zero pad cols 257..287.
// Cols 128..256 are written by gemm1.
__global__ void pack_h_kernel(const float* __restrict__ h,
                              __bf16* __restrict__ zph, __bf16* __restrict__ zpl) {
    int c = threadIdx.x;                      // 0..159
    int r0 = blockIdx.x * ROWS_PER_BLOCK;
    int r1 = r0 + ROWS_PER_BLOCK; if (r1 > NN) r1 = NN;
    if (c < DD) {
        for (int r = r0; r < r1; ++r) {
            float v = h[(size_t)r * DD + c];
            __bf16 hi = (__bf16)v;
            size_t idx = (size_t)r * K2P + c;
            zph[idx] = hi;
            zpl[idx] = (__bf16)(v - (float)hi);
        }
    } else if (c - DD < (K2P - OUTD)) {        // pad cols 257..287
        int pc = OUTD + (c - DD);
        for (int r = r0; r < r1; ++r) {
            size_t idx = (size_t)r * K2P + pc;
            zph[idx] = (__bf16)0.0f;
            zpl[idx] = (__bf16)0.0f;
        }
    }
}

// Weight pack: rows x cols padded, zero fill. One block row per weight row.
__global__ void pack_w_kernel(const float* __restrict__ w, int wr, int wc,
                              __bf16* __restrict__ wph, __bf16* __restrict__ wpl,
                              int pr, int pc) {
    int row = blockIdx.x;                     // 0..pr-1
    for (int c = threadIdx.x; c < pc; c += blockDim.x) {
        float v = (row < wr && c < wc) ? w[(size_t)row * wc + c] : 0.0f;
        __bf16 hi = (__bf16)v;
        size_t idx = (size_t)row * pc + c;
        wph[idx] = hi;
        wpl[idx] = (__bf16)(v - (float)hi);
    }
}

// ---------------------------------------------------------------- WMMA GEMMs
// bf16 hi/lo split: x*w ~= xh*wh + xh*wl + xl*wh (3 WMMAs), f32 accumulate.
// A (16x32): lane M = lane%16; halves at K = kc+8*(lane/16) + {0..7, 16..23}
//            -> two contiguous 8-element (16B) loads.
// B (32x16): lane N = lane%16; halves at K = kc+8*(lane/16) + {0..7, 16..23}
//            -> two contiguous 8-element (16B) loads.
// C/D: lane col = lane%16, VGPR j -> row j + 8*(lane/16).
// Each wave computes a 16x64 strip (NT=4 tiles), reusing the A fragment.

#define NT 4

// h1 = relu(x @ W1^T + b1); writes split-bf16 result into zp cols 128..256.
__global__ void gemm1_kernel(const __bf16* __restrict__ xph, const __bf16* __restrict__ xpl,
                             const __bf16* __restrict__ wph, const __bf16* __restrict__ wpl,
                             const float* __restrict__ b1,
                             __bf16* __restrict__ zph, __bf16* __restrict__ zpl) {
    int tm = blockIdx.x * 4 + (threadIdx.x >> 5);
    if (tm >= MTILES) return;
    int tg = blockIdx.y;                      // N group 0..2
    int l  = threadIdx.x & 31;
    int hg = l >> 4;
    int mr = l & 15;
    size_t arow = (size_t)(tm * 16 + mr) * K1P;

    v8f acc[NT] = {};
    for (int kc = 0; kc < K1P; kc += 32) {
        int ka = kc + 8 * hg;
        v16bf ah = cat8(*(const v8bf*)(xph + arow + ka), *(const v8bf*)(xph + arow + ka + 16));
        v16bf al = cat8(*(const v8bf*)(xpl + arow + ka), *(const v8bf*)(xpl + arow + ka + 16));
#pragma unroll
        for (int nt = 0; nt < NT; ++nt) {
            int gn = (tg * NT + nt) * 16 + mr;            // < W1ROWS
            size_t boff = (size_t)gn * K1P + kc + 8 * hg;
            v16bf bh = cat8(*(const v8bf*)(wph + boff), *(const v8bf*)(wph + boff + 16));
            v16bf bl = cat8(*(const v8bf*)(wpl + boff), *(const v8bf*)(wpl + boff + 16));
            acc[nt] = __builtin_amdgcn_wmma_f32_16x16x32_bf16(false, ah, false, bh, (short)0, acc[nt], false, false);
            acc[nt] = __builtin_amdgcn_wmma_f32_16x16x32_bf16(false, ah, false, bl, (short)0, acc[nt], false, false);
            acc[nt] = __builtin_amdgcn_wmma_f32_16x16x32_bf16(false, al, false, bh, (short)0, acc[nt], false, false);
        }
    }
#pragma unroll
    for (int nt = 0; nt < NT; ++nt) {
        int gn = (tg * NT + nt) * 16 + mr;
        if (gn < AGGD) {
            float bias = b1[gn];
#pragma unroll
            for (int j = 0; j < 8; ++j) {
                int row = tm * 16 + j + 8 * hg;
                float v = acc[nt][j] + bias;
                v = v > 0.0f ? v : 0.0f;
                __bf16 hi = (__bf16)v;
                size_t idx = (size_t)row * K2P + DD + gn;
                zph[idx] = hi;
                zpl[idx] = (__bf16)(v - (float)hi);
            }
        }
    }
}

// out = Z @ W2^T + b2 (Z = [h | h1], K padded to 288)
__global__ void gemm2_kernel(const __bf16* __restrict__ zph, const __bf16* __restrict__ zpl,
                             const __bf16* __restrict__ wph, const __bf16* __restrict__ wpl,
                             const float* __restrict__ b2,
                             float* __restrict__ out) {
    int tm = blockIdx.x * 4 + (threadIdx.x >> 5);
    if (tm >= MTILES) return;
    int tg = blockIdx.y;                      // N group 0..4
    int l  = threadIdx.x & 31;
    int hg = l >> 4;
    int mr = l & 15;
    size_t arow = (size_t)(tm * 16 + mr) * K2P;

    v8f acc[NT] = {};
    for (int kc = 0; kc < K2P; kc += 32) {
        int ka = kc + 8 * hg;
        v16bf ah = cat8(*(const v8bf*)(zph + arow + ka), *(const v8bf*)(zph + arow + ka + 16));
        v16bf al = cat8(*(const v8bf*)(zpl + arow + ka), *(const v8bf*)(zpl + arow + ka + 16));
#pragma unroll
        for (int nt = 0; nt < NT; ++nt) {
            int gn = (tg * NT + nt) * 16 + mr;            // < W2ROWS
            size_t boff = (size_t)gn * K2P + kc + 8 * hg;
            v16bf bh = cat8(*(const v8bf*)(wph + boff), *(const v8bf*)(wph + boff + 16));
            v16bf bl = cat8(*(const v8bf*)(wpl + boff), *(const v8bf*)(wpl + boff + 16));
            acc[nt] = __builtin_amdgcn_wmma_f32_16x16x32_bf16(false, ah, false, bh, (short)0, acc[nt], false, false);
            acc[nt] = __builtin_amdgcn_wmma_f32_16x16x32_bf16(false, ah, false, bl, (short)0, acc[nt], false, false);
            acc[nt] = __builtin_amdgcn_wmma_f32_16x16x32_bf16(false, al, false, bh, (short)0, acc[nt], false, false);
        }
    }
#pragma unroll
    for (int nt = 0; nt < NT; ++nt) {
        int gn = (tg * NT + nt) * 16 + mr;
        if (gn < OUTD) {
            float bias = b2[gn];
#pragma unroll
            for (int j = 0; j < 8; ++j) {
                int row = tm * 16 + j + 8 * hg;
                out[(size_t)row * OUTD + gn] = acc[nt][j] + bias;
            }
        }
    }
}

// ---------------------------------------------------------------- launch
extern "C" void kernel_launch(void* const* d_in, const int* in_sizes, int n_in,
                              void* d_out, int out_size, void* d_ws, size_t ws_size,
                              hipStream_t stream) {
    const int*   u   = (const int*)d_in[0];
    const int*   v   = (const int*)d_in[1];
    const float* g   = (const float*)d_in[2];
    const float* h   = (const float*)d_in[3];
    // d_in[4] = event scalar (== EE here)
    const float* ru  = (const float*)d_in[5];
    const float* rv  = (const float*)d_in[6];
    const float* bnw = (const float*)d_in[7];
    const float* bnb = (const float*)d_in[8];
    const float* w1w = (const float*)d_in[9];
    const float* w1b = (const float*)d_in[10];
    const float* w2w = (const float*)d_in[11];
    const float* w2b = (const float*)d_in[12];
    float* out = (float*)d_out;

    // ---- workspace layout (bytes, 256-aligned blocks) ----
    char* p = (char*)d_ws;
    auto take = [&](size_t bytes) { char* q = p; p += (bytes + 255) & ~(size_t)255; return q; };
    float*  agg   = (float*)take((size_t)NN * AGGD * 4);
    float*  stats = (float*)take(4 * AGGD * 4);
    __bf16* xph   = (__bf16*)take((size_t)NN * K1P * 2);
    __bf16* xpl   = (__bf16*)take((size_t)NN * K1P * 2);
    __bf16* zph   = (__bf16*)take((size_t)NN * K2P * 2);
    __bf16* zpl   = (__bf16*)take((size_t)NN * K2P * 2);
    __bf16* w1ph  = (__bf16*)take((size_t)W1ROWS * K1P * 2);
    __bf16* w1pl  = (__bf16*)take((size_t)W1ROWS * K1P * 2);
    __bf16* w2ph  = (__bf16*)take((size_t)W2ROWS * K2P * 2);
    __bf16* w2pl  = (__bf16*)take((size_t)W2ROWS * K2P * 2);
    float* sum   = stats;
    float* sumsq = stats + AGGD;
    float* scale = stats + 2 * AGGD;
    float* shift = stats + 3 * AGGD;

    zero_kernel<<<1024, 256, 0, stream>>>(agg, (long long)NN * AGGD);
    zero_kernel<<<1, 256, 0, stream>>>(sum, 2 * AGGD);

    scatter_kernel<<<EE / 8, 256, 0, stream>>>(u, v, g, ru, rv, agg);

    colstats_kernel<<<STAT_BLOCKS, 160, 0, stream>>>(agg, sum, sumsq);
    finalize_kernel<<<1, 160, 0, stream>>>(sum, sumsq, bnw, bnb, scale, shift);

    bn_pack_kernel<<<STAT_BLOCKS, K1P, 0, stream>>>(agg, scale, shift, xph, xpl);
    pack_h_kernel<<<STAT_BLOCKS, 160, 0, stream>>>(h, zph, zpl);
    pack_w_kernel<<<W1ROWS, 160, 0, stream>>>(w1w, AGGD, AGGD, w1ph, w1pl, W1ROWS, K1P);
    pack_w_kernel<<<W2ROWS, 288, 0, stream>>>(w2w, OUTD, OUTD, w2ph, w2pl, W2ROWS, K2P);

    gemm1_kernel<<<dim3((MTILES + 3) / 4, W1ROWS / 64), 128, 0, stream>>>(
        xph, xpl, w1ph, w1pl, w1b, zph, zpl);
    gemm2_kernel<<<dim3((MTILES + 3) / 4, W2ROWS / 64), 128, 0, stream>>>(
        zph, zpl, w2ph, w2pl, w2b, out);
}
